// DynaRandGrads_64115271794989
// MI455X (gfx1250) — matrix operand order repaired and checked by reading, not compiled, not run on hardware
//
#include <hip/hip_runtime.h>
#include <hip/hip_bf16.h>

typedef __attribute__((ext_vector_type(16))) __bf16 v16bf;
typedef __attribute__((ext_vector_type(8)))  float  v8f;

#define BB   16
#define CC   16
#define HH   256
#define WW   256
#define LATD 512
#define KSZ  2304   // C*C*9
#define OUT1 2320   // KSZ + C

// ---------------------------------------------------------------------------
// helpers
// ---------------------------------------------------------------------------
static __device__ inline unsigned pack_bf2(float a, float b) {
    union { unsigned short us[2]; unsigned u; } r;
    __bf16 ba = (__bf16)a, bb = (__bf16)b;
    r.us[0] = __builtin_bit_cast(unsigned short, ba);
    r.us[1] = __builtin_bit_cast(unsigned short, bb);
    return r.u;
}

// ---------------------------------------------------------------------------
// Generic M=16 GEMM:  out = act(A(16xK) @ W(KxN) + bias [+ A2(16xK2)@W2 + bias2])
// one wave per 16x16 output tile, bf16 WMMA accumulation in f32
// ---------------------------------------------------------------------------
static __device__ inline v8f gemm_acc(v8f acc, const float* __restrict__ A,
                                      const float* __restrict__ W,
                                      int K, int N, int m, int n, int half) {
    for (int kk = 0; kk < K; kk += 32) {
        v16bf a, bf;
#pragma unroll
        for (int j = 0; j < 8; ++j) {
            // A layout (16-bit A 16x32): lane m, VGPR j -> K = base+half*8+2(j&3)
            int k0 = kk + ((j >= 4) ? 16 : 0) + (half << 3) + ((j & 3) << 1);
            float2 av = *(const float2*)(A + m * K + k0);
            a[2 * j]     = (__bf16)av.x;
            a[2 * j + 1] = (__bf16)av.y;
            // B layout (16-bit B 32x16): lane n, same K as A: K = base+half*8+2(j&3)
            int kb = kk + ((j >= 4) ? 16 : 0) + (half << 3) + ((j & 3) << 1);
            bf[2 * j]     = (__bf16)W[kb * N + n];
            bf[2 * j + 1] = (__bf16)W[(kb + 1) * N + n];
        }
        acc = __builtin_amdgcn_wmma_f32_16x16x32_bf16(false, a, false, bf,
                                                      (short)0, acc, false, false);
    }
    return acc;
}

__global__ __launch_bounds__(128) void gemm16_wmma(
    const float* __restrict__ A,  const float* __restrict__ W,  const float* __restrict__ bias,
    const float* __restrict__ A2, const float* __restrict__ W2, const float* __restrict__ bias2,
    float* __restrict__ outp, int K, int K2, int N, int leaky) {
    int lane = threadIdx.x & 31;
    int wave = threadIdx.x >> 5;
    int tile = blockIdx.x * 4 + wave;
    if (tile * 16 >= N) return;            // wave-uniform exit
    int half = lane >> 4;
    int m = lane & 15;
    int n = tile * 16 + (lane & 15);

    v8f acc = {};
    acc = gemm_acc(acc, A, W, K, N, m, n, half);
    if (K2 > 0) acc = gemm_acc(acc, A2, W2, K2, N, m, n, half);

    float bsum = bias[n] + (bias2 ? bias2[n] : 0.0f);
#pragma unroll
    for (int j = 0; j < 8; ++j) {
        float v = acc[j] + bsum;           // C/D: lane n, VGPR j -> M = j + 8*half
        if (leaky) v = (v > 0.0f) ? v : 0.2f * v;
        outp[(j + (half << 3)) * N + n] = v;
    }
}

// ---------------------------------------------------------------------------
// Rotate 3x3 filters by theta (bilinear) and pack directly into the
// per-lane WMMA A-fragment layout: apack[b][s(5)][lane(32)][j(8)] dwords,
// K ordered as k = tap*16 + ci  (tap = kh*3+kw, padded to 10 taps)
// ---------------------------------------------------------------------------
static __device__ inline float rot_sample(const float* __restrict__ kb,
                                          int co, int ci, int y, int x,
                                          float c, float s) {
    float gx0 = ((float)x + 0.5f) * (2.0f / 3.0f) - 1.0f;
    float gy0 = ((float)y + 0.5f) * (2.0f / 3.0f) - 1.0f;
    float gx = c * gx0 - s * gy0;
    float gy = s * gx0 + c * gy0;
    float ix = ((gx + 1.0f) * 3.0f - 1.0f) * 0.5f;
    float iy = ((gy + 1.0f) * 3.0f - 1.0f) * 0.5f;
    float ix0f = floorf(ix), iy0f = floorf(iy);
    float wx = ix - ix0f, wy = iy - iy0f;
    int ix0 = (int)ix0f, iy0 = (int)iy0f;
    const float* wbase = kb + (co * 16 + ci) * 9;
    float acc = 0.0f;
#pragma unroll
    for (int dy = 0; dy <= 1; ++dy) {
#pragma unroll
        for (int dx = 0; dx <= 1; ++dx) {
            int xi = ix0 + dx, yi = iy0 + dy;
            bool valid = ((unsigned)xi < 3u) && ((unsigned)yi < 3u);
            int xc = min(max(xi, 0), 2), yc = min(max(yi, 0), 2);
            float val = valid ? wbase[yc * 3 + xc] : 0.0f;
            float wgt = (dx ? wx : 1.0f - wx) * (dy ? wy : 1.0f - wy);
            acc += wgt * val;
        }
    }
    return acc;
}

__global__ __launch_bounds__(256) void rotate_pack(const float* __restrict__ kbuf,
                                                   unsigned* __restrict__ apack) {
    int t = blockIdx.x * blockDim.x + threadIdx.x;
    if (t >= BB * 5 * 32) return;
    int b    = t / 160;
    int r    = t - b * 160;
    int s    = r >> 5;
    int lane = r & 31;
    int co   = lane & 15;
    int half = lane >> 4;
    const float* kb = kbuf + b * OUT1;
    float theta = kb[KSZ + co];
    float cth = cosf(theta), sth = sinf(theta);
#pragma unroll
    for (int j = 0; j < 8; ++j) {
        int kg  = 32 * s + (half << 4) + (j << 1);
        int tap = kg >> 4;
        int ci  = kg & 15;
        unsigned pk = 0u;
        if (tap < 9) {
            int yy = tap / 3, xx = tap - yy * 3;
            float v0 = rot_sample(kb, co, ci,     yy, xx, cth, sth);
            float v1 = rot_sample(kb, co, ci + 1, yy, xx, cth, sth);
            pk = pack_bf2(v0, v1);
        }
        apack[t * 8 + j] = pk;
    }
}

// ---------------------------------------------------------------------------
// Copy x -> out channels [0,16)   (out is (B,32,H,W))
// ---------------------------------------------------------------------------
__global__ __launch_bounds__(256) void copy_x(const float4* __restrict__ xin,
                                              float4* __restrict__ outp) {
    int i = blockIdx.x * 256 + threadIdx.x;       // 16*16*65536/4 = 4194304 vec4s
    int b = i >> 18;                              // / 262144
    int rem = i & 262143;
    outp[b * 524288 + rem] = xin[i];
}

// ---------------------------------------------------------------------------
// Conv: per (b, 4-row block) workgroup.
// Y(16 x 256) = W_b(16x160pad) @ X_slab(160x256) per row, via bf16 WMMA.
// LDS holds x slab pre-packed as bf16 pairs: u32 xs[pair(8)][slabrow(6)][col(258)]
//   xs dword = { bf16(x[ci=2p]), bf16(x[ci=2p+1]) }  -> exactly one WMMA B dword.
// 8 waves x 8 tiles (4 rows x 16 col-tiles); A fragments resident across tiles.
// ---------------------------------------------------------------------------
#define SLABR 6
#define SLABC 258
#define PLANE (SLABR * SLABC)            // 1548 dwords per ci-pair plane

__global__ __launch_bounds__(256) void conv_wmma(const float* __restrict__ x,
                                                 const unsigned* __restrict__ apack,
                                                 float* __restrict__ out) {
    __shared__ unsigned xs[8 * PLANE];   // 49,536 B
    int b   = blockIdx.x >> 6;
    int rb  = blockIdx.x & 63;
    int h0  = rb << 2;                   // first of 4 output rows
    int tid = threadIdx.x;

    // ---- stage + convert x slab (rows h0-1 .. h0+4, halo cols, zero pad) ----
    for (int idx = tid; idx < 8 * PLANE; idx += 256) {
        int p   = idx / PLANE;           // ci pair
        int rem = idx - p * PLANE;
        int r   = rem / SLABC;
        int c   = rem - r * SLABC;
        int y   = h0 + r - 1;
        int gx  = c - 1;
        unsigned pk = 0u;
        if ((unsigned)y < 256u && (unsigned)gx < 256u) {
            const float* xp = x + (((b * 16 + 2 * p) * 256 + y) << 8) + gx;
            pk = pack_bf2(xp[0], xp[65536]);   // ci and ci+1 planes
        }
        xs[idx] = pk;
    }
    __syncthreads();

    int lane = tid & 31;
    int wave = tid >> 5;
    int n    = lane & 15;
    int half = lane >> 4;

    // ---- load pre-swizzled A fragments (rotated filters), 5 k-steps ----
    v16bf afr[5];
#pragma unroll
    for (int s = 0; s < 5; ++s) {
        const uint4* ap = (const uint4*)(apack + (((b * 5 + s) * 32 + lane) << 3));
        union { uint4 q[2]; v16bf v; } u;
        u.q[0] = ap[0];
        u.q[1] = ap[1];
        afr[s] = u.v;
    }

#pragma unroll
    for (int tt = 0; tt < 8; ++tt) {
        int tile = wave * 8 + tt;        // 64 tiles: 4 rows x 16 col-tiles
        int row  = tile >> 4;            // 0..3 within block
        int col0 = (tile & 15) << 4;
        v8f acc = {};
#pragma unroll
        for (int s = 0; s < 5; ++s) {
            int  tap = 2 * s + half;     // B layout: lane half selects K block of 16
            bool ok  = tap < 9;
            int  kh  = tap / 3;
            int  dc  = tap - kh * 3;
            int  base = (row + kh) * SLABC + (col0 + n + dc);
            union { unsigned u[8]; v16bf v; } bu;
#pragma unroll
            for (int j = 0; j < 8; ++j) {          // B: VGPR j -> ci pair j
                int idxj = ok ? (j * PLANE + base) : 0;
                unsigned w = xs[idxj];
                bu.u[j] = ok ? w : 0u;
            }
            acc = __builtin_amdgcn_wmma_f32_16x16x32_bf16(false, afr[s], false, bu.v,
                                                          (short)0, acc, false, false);
        }
        // ---- store Y into out channels [16,32) ----
        float* op = out + ((((b * 32 + 16) * 256 + (h0 + row)) << 8) + col0 + n);
#pragma unroll
        for (int j = 0; j < 8; ++j) {
            int m = j + (half << 3);
            op[(long)m << 16] = acc[j];
        }
    }
}

// ---------------------------------------------------------------------------
// launch
// ---------------------------------------------------------------------------
extern "C" void kernel_launch(void* const* d_in, const int* in_sizes, int n_in,
                              void* d_out, int out_size, void* d_ws, size_t ws_size,
                              hipStream_t stream) {
    const float* x       = (const float*)d_in[0];
    const float* lat     = (const float*)d_in[1];
    const float* b0_w0   = (const float*)d_in[2];
    const float* b0_b0   = (const float*)d_in[3];
    const float* b0_w1   = (const float*)d_in[4];
    const float* b0_b1   = (const float*)d_in[5];
    const float* b0_ws   = (const float*)d_in[6];
    const float* b0_bs   = (const float*)d_in[7];
    const float* b1_w0   = (const float*)d_in[8];
    const float* b1_b0   = (const float*)d_in[9];
    const float* b1_w1   = (const float*)d_in[10];
    const float* b1_b1   = (const float*)d_in[11];
    const float* b1_ws   = (const float*)d_in[12];
    const float* b1_bs   = (const float*)d_in[13];
    float* out = (float*)d_out;

    float* wsf   = (float*)d_ws;
    float* t0    = wsf;                         // 16*512
    float* h     = wsf + 8192;                  // 16*512
    float* t1    = wsf + 16384;                 // 16*1024
    float* kout  = wsf + 32768;                 // 16*2320
    unsigned* apack = (unsigned*)(wsf + 69888); // 16*5*32*8 dwords

    // MLP block 0
    gemm16_wmma<<<8, 128, 0, stream>>>(lat, b0_w0, b0_b0, nullptr, nullptr, nullptr,
                                       t0, LATD, 0, LATD, 1);
    gemm16_wmma<<<8, 128, 0, stream>>>(t0, b0_w1, b0_b1, lat, b0_ws, b0_bs,
                                       h, LATD, LATD, LATD, 0);
    // MLP block 1
    gemm16_wmma<<<16, 128, 0, stream>>>(h, b1_w0, b1_b0, nullptr, nullptr, nullptr,
                                        t1, LATD, 0, 2 * LATD, 1);
    gemm16_wmma<<<37, 128, 0, stream>>>(t1, b1_w1, b1_b1, h, b1_ws, b1_bs,
                                        kout, 2 * LATD, LATD, OUT1, 0);
    // rotate filters + pack to WMMA A-fragment layout
    rotate_pack<<<10, 256, 0, stream>>>(kout, apack);
    // concat part 1: copy x into out[:, 0:16]
    copy_x<<<16384, 256, 0, stream>>>((const float4*)x, (float4*)out);
    // concat part 2: dynamic conv into out[:, 16:32]
    conv_wmma<<<BB * 64, 256, 0, stream>>>(x, apack, out);
}
